// STGATModuleWithRAU_23356032156222
// MI455X (gfx1250) — matrix operand order repaired and checked by reading, not compiled, not run on hardware
//
#include <hip/hip_runtime.h>
#include <hip/hip_fp16.h>
#include <math.h>

#define NN      10000      // nodes
#define NE      160000     // edges
#define SEQ     12
#define FIN     32
#define HEADS   4
#define CH      64
#define HC1     (HEADS*CH) // 256
#define HID     256
#define SLOPE   0.2f

typedef __attribute__((ext_vector_type(16))) _Float16 v16h;
typedef __attribute__((ext_vector_type(8)))  float    v8f;

// ---------------------------------------------------------------------------
// utility kernels
// ---------------------------------------------------------------------------
__global__ void fill_f32(float* __restrict__ p, float v, int n) {
    int i = blockIdx.x * blockDim.x + threadIdx.x;
    if (i < n) p[i] = v;
}

// W: [K][N] f32 row-major  ->  Wt: [N][K] f16 row-major (transposed, converted)
__global__ void conv_transpose_f16(const float* __restrict__ W, _Float16* __restrict__ Wt,
                                   int K, int N) {
    int i = blockIdx.x * blockDim.x + threadIdx.x;
    if (i >= K * N) return;
    int n = i / K, k = i - n * K;
    Wt[i] = (_Float16)W[(size_t)k * N + n];
}

// ---------------------------------------------------------------------------
// WMMA GEMM:  C[M][N] = A[M][K](f32, cvt->f16) * Bt[N][K](f16) + bias[N]
// One wave computes a 16x64 output strip: 4 accumulators share one A fragment
// per K-step (4x A reuse, 4 independent v_wmma_f32_16x16x32_f16 in flight).
// M%16==0, N%64==0, K%32==0 guaranteed by caller.
// ---------------------------------------------------------------------------
__global__ void gemm_wmma_bias(const float* __restrict__ A, const _Float16* __restrict__ Bt,
                               const float* __restrict__ bias, float* __restrict__ C,
                               int M, int N, int K) {
    int wave = (int)(threadIdx.x >> 5);
    int lane = (int)(threadIdx.x & 31);
    int tile = blockIdx.x * (int)(blockDim.x >> 5) + wave;
    int tnc  = N >> 6;                      // 64-wide column strips
    int tm   = tile / tnc;
    int tn   = tile - tm * tnc;
    if (tm * 16 >= M) return;               // wave-uniform: EXEC stays all-ones

    int l15 = lane & 15;
    int hi  = lane >> 4;                    // 0 | 1
    const float*    arow = A  + (size_t)(tm * 16 + l15) * K;
    const _Float16* bp0  = Bt + (size_t)(tn * 64 +  0 + l15) * K;
    const _Float16* bp1  = Bt + (size_t)(tn * 64 + 16 + l15) * K;
    const _Float16* bp2  = Bt + (size_t)(tn * 64 + 32 + l15) * K;
    const _Float16* bp3  = Bt + (size_t)(tn * 64 + 48 + l15) * K;

    v8f acc0 = {}, acc1 = {}, acc2 = {}, acc3 = {};
    for (int k0 = 0; k0 < K; k0 += 32) {
        // A fragment: 16 contiguous K halves starting at K + (hi?16:0),
        // matching the B fragment order.
        int ka = k0 + hi * 16;
        float4 x0 = *(const float4*)(arow + ka);
        float4 x1 = *(const float4*)(arow + ka + 4);
        float4 x2 = *(const float4*)(arow + ka + 8);
        float4 x3 = *(const float4*)(arow + ka + 12);
        float af[16] = { x0.x, x0.y, x0.z, x0.w,  x1.x, x1.y, x1.z, x1.w,
                         x2.x, x2.y, x2.z, x2.w,  x3.x, x3.y, x3.z, x3.w };
        v16h a;
        #pragma unroll
        for (int j = 0; j < 16; ++j) a[j] = (_Float16)af[j];

        int kb = k0 + hi * 16;              // B fragment: 16 contiguous K halves
        v16h b0 = *(const v16h*)(bp0 + kb);
        v16h b1 = *(const v16h*)(bp1 + kb);
        v16h b2 = *(const v16h*)(bp2 + kb);
        v16h b3 = *(const v16h*)(bp3 + kb);

        acc0 = __builtin_amdgcn_wmma_f32_16x16x32_f16(false, a, false, b0, (short)0, acc0, false, false);
        acc1 = __builtin_amdgcn_wmma_f32_16x16x32_f16(false, a, false, b1, (short)0, acc1, false, false);
        acc2 = __builtin_amdgcn_wmma_f32_16x16x32_f16(false, a, false, b2, (short)0, acc2, false, false);
        acc3 = __builtin_amdgcn_wmma_f32_16x16x32_f16(false, a, false, b3, (short)0, acc3, false, false);
    }

    int mb = tm * 16 + hi * 8;
    v8f accs[4] = { acc0, acc1, acc2, acc3 };
    #pragma unroll
    for (int j = 0; j < 4; ++j) {
        int n = tn * 64 + j * 16 + l15;
        float bb = bias[n];
        float* crow = C + (size_t)mb * N + n;
        #pragma unroll
        for (int r = 0; r < 8; ++r) crow[(size_t)r * N] = accs[j][r] + bb;
    }
}

// ---------------------------------------------------------------------------
// GATv2 edge kernels — one wave32 per (edge, head); lanes cover channels
// c = lane, lane+32 so loads/atomics are fully coalesced.
// ---------------------------------------------------------------------------
__device__ __forceinline__ void atomicMaxF(float* addr, float val) {
    unsigned int* ua = (unsigned int*)addr;
    unsigned int old = *ua;
    while (__uint_as_float(old) < val) {
        unsigned int assumed = old;
        old = atomicCAS(ua, assumed, __float_as_uint(val));
        if (old == assumed) break;
    }
}

// logits[e,h] = dot(leaky_relu(xl[src]+xr[dst]+edge_attr@We), att[h]); per-dst max
__global__ void att_logits(const int* __restrict__ ei, const float* __restrict__ eat,
                           const float* __restrict__ We, const float* __restrict__ att,
                           const float* __restrict__ xl, const float* __restrict__ xr,
                           float* __restrict__ logits, float* __restrict__ mbuf,
                           int H, int C) {
    int gid  = blockIdx.x * blockDim.x + threadIdx.x;
    int weh  = gid >> 5;                    // (edge, head) per wave — wave-uniform
    int lane = gid & 31;
    if (weh >= NE * H) return;
    int e = weh / H, h = weh - e * H;
    int s = ei[e], d = ei[NE + e];
    int HC = H * C;
    const float* xls = xl + (size_t)s * HC + h * C;
    const float* xrd = xr + (size_t)d * HC + h * C;
    const float* w0 = We + h * C;
    const float* w1 = We + HC + h * C;
    const float* w2 = We + 2 * HC + h * C;
    const float* at = att + h * C;
    float e0 = eat[e * 3 + 0], e1 = eat[e * 3 + 1], e2 = eat[e * 3 + 2];
    float acc = 0.f;
    for (int c = lane; c < C; c += 32) {    // C=64: c = lane, lane+32
        float v = xls[c] + xrd[c] + e0 * w0[c] + e1 * w1[c] + e2 * w2[c];
        v = v > 0.f ? v : SLOPE * v;
        acc += v * at[c];
    }
    #pragma unroll
    for (int o = 16; o > 0; o >>= 1) acc += __shfl_xor(acc, o, 32);
    if (lane == 0) {
        logits[weh] = acc;
        atomicMaxF(mbuf + (size_t)d * H + h, acc);
    }
}

// p = exp(logit - m[dst]); ssum[dst] += p   (one thread per edge-head)
__global__ void att_exp(const int* __restrict__ ei, const float* __restrict__ logits,
                        const float* __restrict__ mbuf, float* __restrict__ p,
                        float* __restrict__ ssum, int H) {
    int idx = blockIdx.x * blockDim.x + threadIdx.x;
    if (idx >= NE * H) return;
    int e = idx / H, h = idx - e * H;
    int d = ei[NE + e];
    float v = __expf(logits[idx] - mbuf[(size_t)d * H + h]);
    p[idx] = v;
    atomicAdd(ssum + (size_t)d * H + h, v);
}

// out[dst] += (p/ssum[dst]) * xl[src]   (one wave per edge-head, coalesced atomics)
__global__ void att_scatter(const int* __restrict__ ei, const float* __restrict__ p,
                            const float* __restrict__ ssum, const float* __restrict__ xl,
                            float* __restrict__ out, int H, int C) {
    int gid  = blockIdx.x * blockDim.x + threadIdx.x;
    int weh  = gid >> 5;
    int lane = gid & 31;
    if (weh >= NE * H) return;
    int e = weh / H, h = weh - e * H;
    int s = ei[e], d = ei[NE + e];
    int HC = H * C;
    float alpha = p[weh] / (ssum[(size_t)d * H + h] + 1e-16f);
    const float* xls = xl + (size_t)s * HC + h * C;
    float* od = out + (size_t)d * HC + h * C;
    for (int c = lane; c < C; c += 32) atomicAdd(od + c, alpha * xls[c]);
}

// g = elu(agg + bias[col])
__global__ void bias_elu(const float* __restrict__ agg, const float* __restrict__ bias,
                         float* __restrict__ g, int HC, int n) {
    int i = blockIdx.x * blockDim.x + threadIdx.x;
    if (i >= n) return;
    float v = agg[i] + bias[i % HC];
    g[i] = v > 0.f ? v : (__expf(v) - 1.f);
}

// ---------------------------------------------------------------------------
// GRU pointwise update (in-place on h)
// ---------------------------------------------------------------------------
__global__ void gru_update(float* __restrict__ h, const float* __restrict__ gi,
                           const float* __restrict__ gh) {
    int i = blockIdx.x * blockDim.x + threadIdx.x;
    if (i >= NN * HID) return;
    int node = i / HID, j = i - node * HID;
    size_t base = (size_t)node * 3 * HID;
    float r = 1.f / (1.f + __expf(-(gi[base + j]       + gh[base + j])));
    float z = 1.f / (1.f + __expf(-(gi[base + HID + j] + gh[base + HID + j])));
    float n = tanhf(       gi[base + 2 * HID + j] + r * gh[base + 2 * HID + j]);
    float hv = h[i];
    h[i] = (1.f - z) * n + z * hv;
}

// ---------------------------------------------------------------------------
// final pooling + fc
// ---------------------------------------------------------------------------
__global__ void col_mean(const float* __restrict__ h, float* __restrict__ colsum) {
    __shared__ float red[256];
    int col = blockIdx.x;
    float s = 0.f;
    for (int n = threadIdx.x; n < NN; n += blockDim.x) s += h[(size_t)n * HID + col];
    red[threadIdx.x] = s;
    __syncthreads();
    for (int o = 128; o > 0; o >>= 1) {
        if ((int)threadIdx.x < o) red[threadIdx.x] += red[threadIdx.x + o];
        __syncthreads();
    }
    if (threadIdx.x == 0) colsum[col] = red[0] / (float)NN;
}

__global__ void final_fc(const float* __restrict__ colsum, const float* __restrict__ Wfc,
                         const float* __restrict__ bfc, float* __restrict__ out) {
    __shared__ float red[256];
    red[threadIdx.x] = colsum[threadIdx.x] * Wfc[threadIdx.x];
    __syncthreads();
    for (int o = 128; o > 0; o >>= 1) {
        if ((int)threadIdx.x < o) red[threadIdx.x] += red[threadIdx.x + o];
        __syncthreads();
    }
    if (threadIdx.x == 0) out[0] = red[0] + bfc[0];
}

// ---------------------------------------------------------------------------
// host driver
// ---------------------------------------------------------------------------
extern "C" void kernel_launch(void* const* d_in, const int* in_sizes, int n_in,
                              void* d_out, int out_size, void* d_ws, size_t ws_size,
                              hipStream_t stream) {
    const float* X    = (const float*)d_in[0];
    const int*   ei   = (const int*)  d_in[1];
    const float* eat  = (const float*)d_in[2];
    const float* W1l  = (const float*)d_in[3];  const float* b1l  = (const float*)d_in[4];
    const float* W1r  = (const float*)d_in[5];  const float* b1r  = (const float*)d_in[6];
    const float* W1e  = (const float*)d_in[7];  const float* att1 = (const float*)d_in[8];
    const float* bias1= (const float*)d_in[9];
    const float* W2l  = (const float*)d_in[10]; const float* b2l  = (const float*)d_in[11];
    const float* W2r  = (const float*)d_in[12]; const float* b2r  = (const float*)d_in[13];
    const float* W2e  = (const float*)d_in[14]; const float* att2 = (const float*)d_in[15];
    const float* bias2= (const float*)d_in[16];
    const float* Wih  = (const float*)d_in[17]; const float* bih  = (const float*)d_in[18];
    const float* Whh  = (const float*)d_in[19]; const float* bhh  = (const float*)d_in[20];
    const float* Wfc  = (const float*)d_in[21]; const float* bfc  = (const float*)d_in[22];
    float* out = (float*)d_out;

    // ---- workspace carve-out ----
    char* w = (char*)d_ws;
    auto alloc = [&](size_t bytes) -> char* {
        char* p = w; w += (bytes + 255) & ~(size_t)255; return p;
    };
    _Float16* W1lT = (_Float16*)alloc((size_t)HC1 * FIN  * 2);
    _Float16* W1rT = (_Float16*)alloc((size_t)HC1 * FIN  * 2);
    _Float16* W2lT = (_Float16*)alloc((size_t)CH  * HC1  * 2);
    _Float16* W2rT = (_Float16*)alloc((size_t)CH  * HC1  * 2);
    _Float16* WihT = (_Float16*)alloc((size_t)3 * HID * CH  * 2);
    _Float16* WhhT = (_Float16*)alloc((size_t)3 * HID * HID * 2);
    float* xl1  = (float*)alloc((size_t)NN * HC1 * 4);
    float* xr1  = (float*)alloc((size_t)NN * HC1 * 4);
    float* agg1 = (float*)alloc((size_t)NN * HC1 * 4);
    float* g1   = (float*)alloc((size_t)NN * HC1 * 4);
    float* xl2  = (float*)alloc((size_t)NN * CH * 4);
    float* xr2  = (float*)alloc((size_t)NN * CH * 4);
    float* agg2 = (float*)alloc((size_t)NN * CH * 4);
    float* g2   = (float*)alloc((size_t)NN * CH * 4);
    float* gi   = (float*)alloc((size_t)NN * 3 * HID * 4);
    float* gh   = (float*)alloc((size_t)NN * 3 * HID * 4);
    float* hbuf = (float*)alloc((size_t)NN * HID * 4);
    float* logits = (float*)alloc((size_t)NE * HEADS * 4);
    float* pbuf   = (float*)alloc((size_t)NE * HEADS * 4);
    float* mbuf   = (float*)alloc((size_t)NN * HEADS * 4);
    float* sbuf   = (float*)alloc((size_t)NN * HEADS * 4);
    float* colsum = (float*)alloc(256 * 4);

    const int TB = 256;
    auto blocks = [&](long long n) { return (int)((n + TB - 1) / TB); };
    auto gemm = [&](const float* A, const _Float16* Bt, const float* bias, float* C,
                    int M, int N, int K) {
        int tiles = (M / 16) * (N / 64);          // 16x64 strip per wave
        int nb = (tiles + 7) / 8;                 // 8 waves (256 threads) per block
        hipLaunchKernelGGL(gemm_wmma_bias, dim3(nb), dim3(256), 0, stream,
                           A, Bt, bias, C, M, N, K);
    };
    auto fillk = [&](float* p, float v, int n) {
        hipLaunchKernelGGL(fill_f32, dim3(blocks(n)), dim3(TB), 0, stream, p, v, n);
    };

    // ---- weight convert + transpose to f16 [N][K] ----
    hipLaunchKernelGGL(conv_transpose_f16, dim3(blocks(FIN*HC1)),   dim3(TB), 0, stream, W1l, W1lT, FIN, HC1);
    hipLaunchKernelGGL(conv_transpose_f16, dim3(blocks(FIN*HC1)),   dim3(TB), 0, stream, W1r, W1rT, FIN, HC1);
    hipLaunchKernelGGL(conv_transpose_f16, dim3(blocks(HC1*CH)),    dim3(TB), 0, stream, W2l, W2lT, HC1, CH);
    hipLaunchKernelGGL(conv_transpose_f16, dim3(blocks(HC1*CH)),    dim3(TB), 0, stream, W2r, W2rT, HC1, CH);
    hipLaunchKernelGGL(conv_transpose_f16, dim3(blocks(CH*3*HID)),  dim3(TB), 0, stream, Wih, WihT, CH, 3*HID);
    hipLaunchKernelGGL(conv_transpose_f16, dim3(blocks(HID*3*HID)), dim3(TB), 0, stream, Whh, WhhT, HID, 3*HID);

    fillk(hbuf, 0.f, NN * HID);   // h0 = 0

    for (int t = 0; t < SEQ; ++t) {
        const float* Xt = X + (size_t)t * NN * FIN;

        // ---- GATv2 layer 1 ----
        gemm(Xt, W1lT, b1l, xl1, NN, HC1, FIN);
        gemm(Xt, W1rT, b1r, xr1, NN, HC1, FIN);
        fillk(mbuf, -1e30f, NN * HEADS);
        fillk(sbuf, 0.f,    NN * HEADS);
        fillk(agg1, 0.f,    NN * HC1);
        hipLaunchKernelGGL(att_logits,  dim3(blocks((long long)NE*HEADS*32)), dim3(TB), 0, stream,
                           ei, eat, W1e, att1, xl1, xr1, logits, mbuf, HEADS, CH);
        hipLaunchKernelGGL(att_exp,     dim3(blocks(NE*HEADS)), dim3(TB), 0, stream,
                           ei, logits, mbuf, pbuf, sbuf, HEADS);
        hipLaunchKernelGGL(att_scatter, dim3(blocks((long long)NE*HEADS*32)), dim3(TB), 0, stream,
                           ei, pbuf, sbuf, xl1, agg1, HEADS, CH);
        hipLaunchKernelGGL(bias_elu,    dim3(blocks(NN*HC1)), dim3(TB), 0, stream,
                           agg1, bias1, g1, HC1, NN * HC1);

        // ---- GATv2 layer 2 (1 head) ----
        gemm(g1, W2lT, b2l, xl2, NN, CH, HC1);
        gemm(g1, W2rT, b2r, xr2, NN, CH, HC1);
        fillk(mbuf, -1e30f, NN);
        fillk(sbuf, 0.f,    NN);
        fillk(agg2, 0.f,    NN * CH);
        hipLaunchKernelGGL(att_logits,  dim3(blocks((long long)NE*32)), dim3(TB), 0, stream,
                           ei, eat, W2e, att2, xl2, xr2, logits, mbuf, 1, CH);
        hipLaunchKernelGGL(att_exp,     dim3(blocks(NE)), dim3(TB), 0, stream,
                           ei, logits, mbuf, pbuf, sbuf, 1);
        hipLaunchKernelGGL(att_scatter, dim3(blocks((long long)NE*32)), dim3(TB), 0, stream,
                           ei, pbuf, sbuf, xl2, agg2, 1, CH);
        hipLaunchKernelGGL(bias_elu,    dim3(blocks(NN*CH)), dim3(TB), 0, stream,
                           agg2, bias2, g2, CH, NN * CH);

        // ---- RAU / GRU cell ----
        gemm(g2,   WihT, bih, gi, NN, 3 * HID, CH);
        gemm(hbuf, WhhT, bhh, gh, NN, 3 * HID, HID);
        hipLaunchKernelGGL(gru_update, dim3(blocks(NN*HID)), dim3(TB), 0, stream,
                           hbuf, gi, gh);
    }

    // ---- global mean pool + fc ----
    hipLaunchKernelGGL(col_mean, dim3(HID), dim3(256), 0, stream, hbuf, colsum);
    hipLaunchKernelGGL(final_fc, dim3(1),   dim3(256), 0, stream, colsum, Wfc, bfc, out);
}
